// MultiHeadedAttention_38465727103377
// MI455X (gfx1250) — compile-verified, matrix operand order checked by reading, not run on hardware
//
#include <hip/hip_runtime.h>
#include <hip/hip_bf16.h>

// MI455X / gfx1250 fused multi-head attention, bf16 WMMA (16x16x32) everywhere.
// Pipeline: mask bit-pack -> 3x projection GEMM -> fused flash attention
//           (async LDS staging) -> output GEMM.

typedef __attribute__((ext_vector_type(16))) __bf16 v16bf;
typedef __attribute__((ext_vector_type(8)))  float  v8f;
typedef __attribute__((ext_vector_type(4)))  int    v4i;

#define BATCH   4
#define L_SEQ   2048
#define DMODEL  1024
#define NHEAD   16
#define DHEAD   64
#define NEGVAL  (-1e18f)
#define LWORDS  (L_SEQ / 32)   // packed mask words per row = 64

#if __has_builtin(__builtin_amdgcn_global_load_async_to_lds_b128)
#define HAVE_ASYNC_LDS 1
#else
#define HAVE_ASYNC_LDS 0
#endif

__device__ __forceinline__ v8f v8f_zero() {
  v8f z;
#pragma unroll
  for (int i = 0; i < 8; ++i) z[i] = 0.0f;
  return z;
}

// K-index pattern for 16-bit A/B fragments (ISA 7.12.2, 16-bit A 16x32):
// lanes 0-15 hold K {0..7,16..23}, lanes 16-31 hold K {8..15,24..31}.
__device__ __forceinline__ int kpat(int i, int half) {
  return ((i >> 3) << 4) + (half << 3) + (i & 7);
}

// A-style fragment (16 rows x 32 K) from a row-major [row][k] bf16 tile.
// Also used for B fragments when the source tile is stored [n][k] (i.e. K^T).
__device__ __forceinline__ v16bf load_frag_rowmajor(const __bf16* base, int ld,
                                                    int row0, int k0, int lane) {
  const int r = row0 + (lane & 15);
  const int h = lane >> 4;
  v16bf f;
#pragma unroll
  for (int i = 0; i < 16; ++i) f[i] = base[r * ld + k0 + kpat(i, h)];
  return f;
}

// B-style fragment (32 K x 16 cols) from a k-major [k][n] bf16 tile.
__device__ __forceinline__ v16bf load_frag_kmajor(const __bf16* base, int ld,
                                                  int k0, int col0, int lane) {
  const int n = col0 + (lane & 15);
  const int h = lane >> 4;
  v16bf f;
#pragma unroll
  for (int i = 0; i < 16; ++i) f[i] = base[(k0 + kpat(i, h)) * ld + n];
  return f;
}

__device__ __forceinline__ v8f wmma_bf16(v16bf a, v16bf b, v8f c) {
  return __builtin_amdgcn_wmma_f32_16x16x32_bf16(false, a, false, b,
                                                 (short)0, c, false, false);
}

__device__ __forceinline__ float redmax16(float v) {
  v = fmaxf(v, __shfl_xor(v, 1));
  v = fmaxf(v, __shfl_xor(v, 2));
  v = fmaxf(v, __shfl_xor(v, 4));
  v = fmaxf(v, __shfl_xor(v, 8));
  return v;
}
__device__ __forceinline__ float redsum16(float v) {
  v += __shfl_xor(v, 1);
  v += __shfl_xor(v, 2);
  v += __shfl_xor(v, 4);
  v += __shfl_xor(v, 8);
  return v;
}

// One 16-byte global->LDS tile stage: async DMA on gfx1250 (ASYNCcnt path),
// else a plain b128 load + ds_store. The builtin's params are v4i pointers in
// explicit address spaces (per the round-2 diagnostic), so cast stepwise:
// const T* -> void* -> AS void* -> AS v4i*.
__device__ __forceinline__ void stage_b128(const __bf16* gsrc, __bf16* ldst) {
#if HAVE_ASYNC_LDS
  __builtin_amdgcn_global_load_async_to_lds_b128(
      (__attribute__((address_space(1))) v4i*)
          (__attribute__((address_space(1))) void*)(void*)(const void*)gsrc,
      (__attribute__((address_space(3))) v4i*)
          (__attribute__((address_space(3))) void*)(void*)ldst,
      0, 0);
#else
  *(uint4*)ldst = *(const uint4*)gsrc;
#endif
}

__device__ __forceinline__ void stage_wait() {
#if HAVE_ASYNC_LDS
#if __has_builtin(__builtin_amdgcn_s_wait_asynccnt)
  __builtin_amdgcn_s_wait_asynccnt(0);
#else
  asm volatile("s_wait_asynccnt 0" ::: "memory");
#endif
#endif
}

// ---------------------------------------------------------------------------
// Pack the byte mask into bit-words: word w of row q holds keys [32w, 32w+32),
// bit k = mask[q][32w+k]. One thread per word (32 mask bytes -> 1 u32).
// ---------------------------------------------------------------------------
__global__ __launch_bounds__(256) void pack_mask_kernel(
    const unsigned char* __restrict__ mask, unsigned int* __restrict__ pm) {
  const size_t idx = (size_t)blockIdx.x * 256 + threadIdx.x;
  const unsigned char* src = mask + idx * 32;
  const uint4 a = *(const uint4*)(src);
  const uint4 b = *(const uint4*)(src + 16);
  const unsigned int words[8] = {a.x, a.y, a.z, a.w, b.x, b.y, b.z, b.w};
  unsigned int w = 0;
#pragma unroll
  for (int j = 0; j < 8; ++j) {
    const unsigned int v = words[j];
    w |= ((v & 0x000000FFu) ? 1u : 0u) << (j * 4 + 0);
    w |= ((v & 0x0000FF00u) ? 1u : 0u) << (j * 4 + 1);
    w |= ((v & 0x00FF0000u) ? 1u : 0u) << (j * 4 + 2);
    w |= ((v & 0xFF000000u) ? 1u : 0u) << (j * 4 + 3);
  }
  pm[idx] = w;
}

// ---------------------------------------------------------------------------
// GEMM: Out = (X @ W + bias) * scale.  X:[M=8192,K=1024] (TIN), W:[K,N] fp32.
// OUTMODE 0: write bf16 in head-major [b,h,l,dh] (projection).
// OUTMODE 1: write fp32 row-major [m,n] (final output projection).
// Block = 256 threads (8 waves), tile 128x128, K-step 32, wave tile 32x64.
// ---------------------------------------------------------------------------
template <typename TIN, int OUTMODE>
__global__ __launch_bounds__(256) void gemm_kernel(
    const TIN* __restrict__ X, const float* __restrict__ W,
    const float* __restrict__ bias, float scale,
    __bf16* __restrict__ OutBf, float* __restrict__ OutF) {
  constexpr int N  = DMODEL;
  constexpr int Kd = DMODEL;

  __shared__ __bf16 As[128][40];   // 128 x 32 (+8 pad)
  __shared__ __bf16 Bs[32][136];   // 32 x 128 (+8 pad)

  const int tid  = threadIdx.x;
  const int wid  = tid >> 5;
  const int lane = tid & 31;
  const int half = lane >> 4;
  const int nidx = lane & 15;
  const int m0 = blockIdx.y * 128;
  const int n0 = blockIdx.x * 128;
  const int wm = wid >> 1;         // 4 wave-rows of 32
  const int wn = wid & 1;          // 2 wave-cols of 64

  v8f acc[2][4];
#pragma unroll
  for (int mi = 0; mi < 2; ++mi)
#pragma unroll
    for (int ni = 0; ni < 4; ++ni) acc[mi][ni] = v8f_zero();

  const int arow = tid >> 1, acol = (tid & 1) * 16;
  const int brow = tid >> 3, bcol = (tid & 7) * 16;

  for (int k0 = 0; k0 < Kd; k0 += 32) {
#pragma unroll
    for (int j = 0; j < 16; ++j)
      As[arow][acol + j] =
          (__bf16)(float)X[(size_t)(m0 + arow) * Kd + k0 + acol + j];
#pragma unroll
    for (int j = 0; j < 16; ++j)
      Bs[brow][bcol + j] = (__bf16)W[(size_t)(k0 + brow) * N + n0 + bcol + j];
    if (k0 + 32 < Kd) {  // global_prefetch_b8 hint for next K tile
      __builtin_prefetch(&X[(size_t)(m0 + arow) * Kd + k0 + 32 + acol], 0, 0);
      __builtin_prefetch(&W[(size_t)(k0 + 32 + brow) * N + n0 + bcol], 0, 0);
    }
    __syncthreads();

    v16bf a0 = load_frag_rowmajor(&As[0][0], 40, wm * 32, 0, lane);
    v16bf a1 = load_frag_rowmajor(&As[0][0], 40, wm * 32 + 16, 0, lane);
#pragma unroll
    for (int ni = 0; ni < 4; ++ni) {
      v16bf bf = load_frag_kmajor(&Bs[0][0], 136, 0, wn * 64 + ni * 16, lane);
      acc[0][ni] = wmma_bf16(a0, bf, acc[0][ni]);
      acc[1][ni] = wmma_bf16(a1, bf, acc[1][ni]);
    }
    __syncthreads();
  }

#pragma unroll
  for (int mi = 0; mi < 2; ++mi)
#pragma unroll
    for (int ni = 0; ni < 4; ++ni)
#pragma unroll
      for (int r = 0; r < 8; ++r) {
        const int m = m0 + wm * 32 + mi * 16 + r + 8 * half;
        const int n = n0 + wn * 64 + ni * 16 + nidx;
        const float v = (acc[mi][ni][r] + bias[n]) * scale;
        if (OUTMODE == 0) {
          const int bb = m >> 11;          // / L_SEQ
          const int l  = m & (L_SEQ - 1);
          const int hh = n >> 6;           // / DHEAD
          const int d  = n & (DHEAD - 1);
          OutBf[(((size_t)bb * NHEAD + hh) * L_SEQ + l) * DHEAD + d] = (__bf16)v;
        } else {
          OutF[(size_t)m * N + n] = v;
        }
      }
}

// ---------------------------------------------------------------------------
// Fused attention. Grid: (L/128, B*H). Block: 256 thr = 8 waves, each wave
// owns 16 query rows. Online softmax over 32-key tiles; K/V tiles staged via
// async global->LDS DMA. Mask consumed as packed bit-words (1 u32 per row per
// tile). Head 0 recomputes scores to emit exact probabilities with b128
// stores through a per-wave fp32 LDS bounce.
// ---------------------------------------------------------------------------
__global__ __launch_bounds__(256) void attn_kernel(
    const __bf16* __restrict__ Q, const __bf16* __restrict__ K,
    const __bf16* __restrict__ V, const unsigned int* __restrict__ PM,
    __bf16* __restrict__ Ctx, float* __restrict__ TopAttn) {
  __shared__ __bf16 Ks[32][72];        // 32 keys x 64 dh (+8 pad)
  __shared__ __bf16 Vs[32][72];
  __shared__ __bf16 Ps[8][16][40];     // per-wave attn scratch (bf16, A-layout)
  __shared__ float  Pf[8][16][33];     // per-wave fp32 scratch for head-0 out

  const int tid  = threadIdx.x;
  const int wid  = tid >> 5;
  const int lane = tid & 31;
  const int half = lane >> 4;
  const int nidx = lane & 15;

  const int bh = blockIdx.y;
  const int b  = bh / NHEAD;
  const int h  = bh % NHEAD;
  const int q0 = blockIdx.x * 128 + wid * 16;

  const __bf16* Qb = Q + (size_t)bh * L_SEQ * DHEAD;
  const __bf16* Kb = K + (size_t)bh * L_SEQ * DHEAD;
  const __bf16* Vb = V + (size_t)bh * L_SEQ * DHEAD;
  const unsigned int* PMb = PM + (size_t)b * L_SEQ * LWORDS;

  // Q fragments (A layout): rows q0..q0+15, dh split into two 32-wide chunks.
  v16bf qa0, qa1;
  {
    const int r = q0 + nidx;
#pragma unroll
    for (int i = 0; i < 16; ++i) {
      qa0[i] = Qb[(size_t)r * DHEAD + kpat(i, half)];
      qa1[i] = Qb[(size_t)r * DHEAD + 32 + kpat(i, half)];
    }
  }

  float rmax[8], rsum[8];
#pragma unroll
  for (int r = 0; r < 8; ++r) { rmax[r] = -3.0e38f; rsum[r] = 0.0f; }
  v8f cacc[4];
#pragma unroll
  for (int f = 0; f < 4; ++f) cacc[f] = v8f_zero();

  const int ldrow = tid >> 3;          // staging: 16B per thread per tile
  const int ldcol = (tid & 7) * 8;

  for (int kt = 0; kt < L_SEQ / 32; ++kt) {
    stage_b128(&Kb[(size_t)(kt * 32 + ldrow) * DHEAD + ldcol], &Ks[ldrow][ldcol]);
    stage_b128(&Vb[(size_t)(kt * 32 + ldrow) * DHEAD + ldcol], &Vs[ldrow][ldcol]);
    stage_wait();
    __syncthreads();

    // Scores: S = Q (16x64) x K^T; Ks is [key][d] so row-major frag = B^T.
    v8f s0 = v8f_zero(), s1 = v8f_zero();
    {
      v16bf kb;
      kb = load_frag_rowmajor(&Ks[0][0], 72, 0, 0, lane);
      s0 = wmma_bf16(qa0, kb, s0);
      kb = load_frag_rowmajor(&Ks[0][0], 72, 0, 32, lane);
      s0 = wmma_bf16(qa1, kb, s0);
      kb = load_frag_rowmajor(&Ks[0][0], 72, 16, 0, lane);
      s1 = wmma_bf16(qa0, kb, s1);
      kb = load_frag_rowmajor(&Ks[0][0], 72, 16, 32, lane);
      s1 = wmma_bf16(qa1, kb, s1);
    }

    // Mask (packed bits) + online softmax. C layout: lane = key column.
    float al[8], p0[8], p1[8];
#pragma unroll
    for (int r = 0; r < 8; ++r) {
      const int qrow = q0 + r + 8 * half;
      const unsigned int mw = PMb[(size_t)qrow * LWORDS + kt];
      float v0 = s0[r], v1 = s1[r];
      if ((mw >> nidx) & 1u)        v0 = NEGVAL;
      if ((mw >> (nidx + 16)) & 1u) v1 = NEGVAL;
      const float tm = redmax16(fmaxf(v0, v1));
      const float nm = fmaxf(rmax[r], tm);
      al[r] = __expf(rmax[r] - nm);
      p0[r] = __expf(v0 - nm);
      p1[r] = __expf(v1 - nm);
      rsum[r] = rsum[r] * al[r] + redsum16(p0[r] + p1[r]);
      rmax[r] = nm;
    }
#pragma unroll
    for (int f = 0; f < 4; ++f)
#pragma unroll
      for (int r = 0; r < 8; ++r) cacc[f][r] *= al[r];

    // C-layout -> A-layout transpose through per-wave LDS scratch.
#pragma unroll
    for (int r = 0; r < 8; ++r) {
      Ps[wid][r + 8 * half][nidx]      = (__bf16)p0[r];
      Ps[wid][r + 8 * half][16 + nidx] = (__bf16)p1[r];
    }
    asm volatile("s_wait_dscnt 0" ::: "memory");

    const v16bf pa = load_frag_rowmajor(&Ps[wid][0][0], 40, 0, 0, lane);
#pragma unroll
    for (int f = 0; f < 4; ++f) {
      v16bf vb = load_frag_kmajor(&Vs[0][0], 72, 0, f * 16, lane);
      cacc[f] = wmma_bf16(pa, vb, cacc[f]);
    }
    __syncthreads();
  }

  float inv[8];
#pragma unroll
  for (int r = 0; r < 8; ++r) inv[r] = 1.0f / rsum[r];

  // ctx written [b, l, h*dh + d] so the final GEMM reads row-major [M, D].
#pragma unroll
  for (int f = 0; f < 4; ++f)
#pragma unroll
    for (int r = 0; r < 8; ++r) {
      const int qrow = q0 + r + 8 * half;
      const int d    = f * 16 + nidx;
      Ctx[((size_t)b * L_SEQ + qrow) * DMODEL + h * DHEAD + d] =
          (__bf16)(cacc[f][r] * inv[r]);
    }

  // Head 0: recompute scores, emit exact softmax probabilities (fp32).
  if (h == 0) {
    float* Tb = TopAttn + (size_t)b * L_SEQ * L_SEQ;
    for (int kt = 0; kt < L_SEQ / 32; ++kt) {
      stage_b128(&Kb[(size_t)(kt * 32 + ldrow) * DHEAD + ldcol],
                 &Ks[ldrow][ldcol]);
      stage_wait();
      __syncthreads();
      v8f s0 = v8f_zero(), s1 = v8f_zero();
      {
        v16bf kb;
        kb = load_frag_rowmajor(&Ks[0][0], 72, 0, 0, lane);
        s0 = wmma_bf16(qa0, kb, s0);
        kb = load_frag_rowmajor(&Ks[0][0], 72, 0, 32, lane);
        s0 = wmma_bf16(qa1, kb, s0);
        kb = load_frag_rowmajor(&Ks[0][0], 72, 16, 0, lane);
        s1 = wmma_bf16(qa0, kb, s1);
        kb = load_frag_rowmajor(&Ks[0][0], 72, 16, 32, lane);
        s1 = wmma_bf16(qa1, kb, s1);
      }
#pragma unroll
      for (int r = 0; r < 8; ++r) {
        const int qrow = q0 + r + 8 * half;
        const unsigned int mw = PMb[(size_t)qrow * LWORDS + kt];
        float v0 = s0[r], v1 = s1[r];
        if ((mw >> nidx) & 1u)        v0 = NEGVAL;
        if ((mw >> (nidx + 16)) & 1u) v1 = NEGVAL;
        Pf[wid][r + 8 * half][nidx]      = __expf(v0 - rmax[r]) * inv[r];
        Pf[wid][r + 8 * half][16 + nidx] = __expf(v1 - rmax[r]) * inv[r];
      }
      asm volatile("s_wait_dscnt 0" ::: "memory");
      // Lane writes 16 contiguous floats of one query row as 4x b128 stores.
      {
        const int qrow = q0 + nidx;
#pragma unroll
        for (int c = 0; c < 4; ++c) {
          float4 t;
          t.x = Pf[wid][nidx][half * 16 + c * 4 + 0];
          t.y = Pf[wid][nidx][half * 16 + c * 4 + 1];
          t.z = Pf[wid][nidx][half * 16 + c * 4 + 2];
          t.w = Pf[wid][nidx][half * 16 + c * 4 + 3];
          *(float4*)&Tb[(size_t)qrow * L_SEQ + kt * 32 + half * 16 + c * 4] = t;
        }
      }
      __syncthreads();
    }
  }
}

// ---------------------------------------------------------------------------
extern "C" void kernel_launch(void* const* d_in, const int* in_sizes, int n_in,
                              void* d_out, int out_size, void* d_ws,
                              size_t ws_size, hipStream_t stream) {
  const float* key   = (const float*)d_in[0];
  const float* value = (const float*)d_in[1];
  const float* query = (const float*)d_in[2];
  const unsigned char* mask = (const unsigned char*)d_in[3];  // numpy bool
  const float* Wk = (const float*)d_in[4];
  const float* bk = (const float*)d_in[5];
  const float* Wv = (const float*)d_in[6];
  const float* bv = (const float*)d_in[7];
  const float* Wq = (const float*)d_in[8];
  const float* bq = (const float*)d_in[9];
  const float* Wo = (const float*)d_in[10];
  const float* bo = (const float*)d_in[11];

  float* out     = (float*)d_out;
  float* topattn = out + (size_t)BATCH * L_SEQ * DMODEL;

  const size_t tens = (size_t)BATCH * L_SEQ * DMODEL;  // 8,388,608 elems
  __bf16* Kbf = (__bf16*)d_ws;        // bf16 scratch: 4 x 16 MiB
  __bf16* Vbf = Kbf + tens;
  __bf16* Qbf = Vbf + tens;
  __bf16* Ctx = Qbf + tens;
  unsigned int* PMsk = (unsigned int*)(Ctx + tens);    // +2 MiB packed mask

  // Pack mask bits: 4*2048*64 = 524288 words, one thread each.
  pack_mask_kernel<<<dim3((BATCH * L_SEQ * LWORDS) / 256), 256, 0, stream>>>(
      mask, PMsk);

  const dim3 gg(DMODEL / 128, (BATCH * L_SEQ) / 128);  // (8, 64)
  gemm_kernel<float, 0><<<gg, 256, 0, stream>>>(key,   Wk, bk, 1.0f,   Kbf, nullptr);
  gemm_kernel<float, 0><<<gg, 256, 0, stream>>>(value, Wv, bv, 1.0f,   Vbf, nullptr);
  gemm_kernel<float, 0><<<gg, 256, 0, stream>>>(query, Wq, bq, 0.125f, Qbf, nullptr);

  const dim3 ga(L_SEQ / 128, BATCH * NHEAD);           // (16, 64)
  attn_kernel<<<ga, 256, 0, stream>>>(Qbf, Kbf, Vbf, PMsk, Ctx, topattn);

  gemm_kernel<__bf16, 1><<<gg, 256, 0, stream>>>(Ctx, Wo, bo, 1.0f, nullptr, out);
}
